// DINOBevMixedAttentionMHA_56315611185805
// MI455X (gfx1250) — matrix operand order repaired and checked by reading, not compiled, not run on hardware
//
#include <hip/hip_runtime.h>

// ---------------------------------------------------------------------------
// DINO-BEV mixed attention for gfx1250 (MI455X, wave32, WMMA + TDM).
// All activations are stored L-major (transposed) so every WMMA fragment
// load is contiguous-K per lane (b128 global/ds loads).
// d_in layout assumption (setup_inputs dict order, params flattened in order):
//  0 bev_feat(256*2500) 1 last_tokens(6*2816*768) 2 lidar2img(96) 3 z_samples(4)
//  4 norm_q_g 5 norm_q_b 6 q_proj_w 7 q_proj_b 8 self_mlp1_w 9 self_mlp1_b
// 10 self_mlp2_w 11 self_mlp2_b 12 wk_self_w 13 wk_self_b 14 wv_self_w 15 wv_self_b
// 16 dino_mlp1_w 17 dino_mlp1_b 18 dino_mlp2_w 19 dino_mlp2_b 20 dino_c_w 21 dino_c_b
// 22 wk_dino_w 23 wk_dino_b 24 wv_dino_w 25 wv_dino_b 26 proj_out_w 27 proj_out_b
// 28 dino_gate(1)
// ---------------------------------------------------------------------------

typedef _Float16 f16;
typedef __attribute__((ext_vector_type(16))) _Float16 v16h;
typedef __attribute__((ext_vector_type(8)))  float    v8f;

#define L1V    2500
#define LQP    2512            // 157 * 16
#define LKP    5024            // 157 * 32 (keys padded, masked in softmax)
#define LKVAL  5000
#define CDINO  768
#define NHEAD  8
#define DKH    32
#define KTILES 157             // ceil(5000/32)
#define HPV    32
#define WPV    88
#define PATCH  (HPV*WPV)
#define VC     6
#define SZ     4
#define SCALE_QK 0.4204482076268573f   // 32^-0.25

#if defined(__has_builtin)
#  if __has_builtin(__builtin_amdgcn_tensor_load_to_lds) && \
      __has_builtin(__builtin_amdgcn_s_wait_tensorcnt)
#    define HAVE_TDM 1
#  else
#    define HAVE_TDM 0
#  endif
#else
#  define HAVE_TDM 0
#endif

// ---------------- WMMA fragment loaders (wave32 layouts, ISA 7.12.2) --------

// A 16x32 f16, memory row-major (m rows, contiguous K). p -> &A[m][k0]
// Per lane: halves {16*h2..16*h2+15} -> two b128 loads.
__device__ inline v16h frag_a_rm(const f16* __restrict__ p, int h2) {
    v16h a;
#pragma unroll
    for (int j = 0; j < 8; ++j) {
        int K0 = 16 * h2 + 2 * j;
        a[2*j]   = p[K0];
        a[2*j+1] = p[K0+1];
    }
    return a;
}
// B 32x16 f16, memory n-major with contiguous K. p -> &mem[n][k0]
// Per lane: halves {16*h2..16*h2+15} -> two b128 loads.
__device__ inline v16h frag_b_nm(const f16* __restrict__ p, int h2) {
    v16h b;
#pragma unroll
    for (int j = 0; j < 8; ++j) {
        int K0 = 16 * h2 + 2 * j;
        b[2*j]   = p[K0];
        b[2*j+1] = p[K0+1];
    }
    return b;
}

__device__ inline float gelu_exact(float a) {
    return 0.5f * a * (1.0f + erff(a * 0.7071067811865475f));
}

// ---------------- TDM helper: 2D f16 tile (rows x cols) -> LDS --------------

#if HAVE_TDM
typedef __attribute__((ext_vector_type(4))) unsigned tdm_v4u;
typedef __attribute__((ext_vector_type(8))) int      tdm_v8i;
typedef __attribute__((ext_vector_type(4))) int      tdm_v4i;

__device__ inline void tdm_load_tile_f16(unsigned lds_addr, const f16* gptr,
                                         int rows, int cols, int row_stride) {
    unsigned long long ga = (unsigned long long)(size_t)gptr;
    tdm_v4u g0;
    g0[0] = 1u;                                   // count=1 user descriptor
    g0[1] = lds_addr;                             // LDS byte address
    g0[2] = (unsigned)(ga & 0xFFFFFFFFu);         // global_addr[31:0]
    g0[3] = (unsigned)((ga >> 32) & 0x01FFFFFFu)  // global_addr[56:32]
          | (2u << 30);                           // type=2 ("image")
    tdm_v8i g1;
    g1[0] = (int)(1u << 16);                      // data_size=1 (2 bytes)
    g1[1] = (int)((unsigned)cols << 16);          // tensor_dim0[15:0]
    g1[2] = (int)((unsigned)rows << 16);          // tensor_dim1[15:0]
    g1[3] = (int)((unsigned)cols << 16);          // tile_dim0
    g1[4] = rows;                                 // tile_dim1 (tile_dim2=0)
    g1[5] = row_stride;                           // tensor_dim0_stride[31:0]
    g1[6] = 0;
    g1[7] = 0;
    tdm_v4i gz = {0, 0, 0, 0};
#if __clang_major__ >= 23
    tdm_v8i gz8 = {0, 0, 0, 0, 0, 0, 0, 0};
    __builtin_amdgcn_tensor_load_to_lds(g0, g1, gz, gz, gz8, 0);
#else
    __builtin_amdgcn_tensor_load_to_lds(g0, g1, gz, gz, 0);
#endif
}
#endif

// ---------------- utility kernels -------------------------------------------

__global__ __launch_bounds__(256) void zero16_k(f16* __restrict__ p, int n) {
    int i = blockIdx.x * 256 + threadIdx.x;
    if (i < n) p[i] = (f16)0.0f;
}

__global__ __launch_bounds__(256) void cvt16_k(const float* __restrict__ in,
                                               f16* __restrict__ out, int n) {
    int i = blockIdx.x * 256 + threadIdx.x;
    if (i < n) out[i] = (f16)in[i];
}

// (C x L1) f32 -> (L1 x ldo) f16 transposed copy, coalesced stores
__global__ __launch_bounds__(256) void xpose_cvt_k(const float* __restrict__ in,
                                                   f16* __restrict__ out, int ldo) {
    int l = blockIdx.x, c = threadIdx.x;
    out[(size_t)l * ldo + c] = (f16)in[(size_t)c * L1V + l];
}

__global__ void sigm_k(const float* __restrict__ g, float* __restrict__ o) {
    o[0] = 1.0f / (1.0f + expf(-g[0]));
}

// ---------------- GroupNorm(32) -> f16 L-major (q_proj input) ---------------

__global__ __launch_bounds__(256) void gn_k(const float* __restrict__ x,
                                            const float* __restrict__ gw,
                                            const float* __restrict__ gb,
                                            f16* __restrict__ xnT) {
    const int grp = blockIdx.x;
    const int t   = threadIdx.x;
    const int nel = 8 * L1V;
    const float* xg = x + (size_t)grp * 8 * L1V;
    float s1 = 0.0f, s2 = 0.0f;
    for (int e = t; e < nel; e += 256) { float v = xg[e]; s1 += v; s2 += v * v; }
    __shared__ float r1[256], r2[256];
    r1[t] = s1; r2[t] = s2;
    __syncthreads();
    for (int o = 128; o > 0; o >>= 1) {
        if (t < o) { r1[t] += r1[t + o]; r2[t] += r2[t + o]; }
        __syncthreads();
    }
    float mu  = r1[0] * (1.0f / nel);
    float var = r2[0] * (1.0f / nel) - mu * mu;
    float rs  = rsqrtf(var + 1e-5f);
    for (int e = t; e < nel; e += 256) {
        int l = e >> 3, c = e & 7;
        int ch = grp * 8 + c;
        float vn = (xg[(size_t)c * L1V + l] - mu) * rs;
        xnT[(size_t)l * 256 + ch] = (f16)(vn * gw[ch] + gb[ch]);
    }
}

// ---------------- geometry: project pillars, fold masked-mean coefs ---------

__global__ __launch_bounds__(256) void geom_k(const float* __restrict__ l2i,
                                              const float* __restrict__ zs,
                                              int* __restrict__ midx,
                                              float* __restrict__ mw,
                                              float* __restrict__ posf) {
    __shared__ float sM[VC * 16];
    __shared__ float sZ[SZ];
    int t = threadIdx.x;
    if (t < VC * 16) sM[t] = l2i[t];
    if (t < SZ)      sZ[t] = zs[t];
    __syncthreads();
    int l = blockIdx.x * 256 + t;
    if (l >= L1V) return;
    int i = l / 50, j = l % 50;
    float bx = -51.2f + j * (102.4f / 49.0f);
    float by = -51.2f + i * (102.4f / 49.0f);

    float cnt[SZ] = {0.f, 0.f, 0.f, 0.f};
    unsigned viewany = 0u;
    for (int v = 0; v < VC; ++v) {
        const float* M = sM + v * 16;
        for (int s = 0; s < SZ; ++s) {
            float z3 = sZ[s];
            float c0 = M[0]*bx + M[1]*by + M[2]*z3  + M[3];
            float c1 = M[4]*bx + M[5]*by + M[6]*z3  + M[7];
            float c2 = M[8]*bx + M[9]*by + M[10]*z3 + M[11];
            float zz = fmaxf(c2, 1e-6f);
            float u = c0 / zz, vv = c1 / zz;
            bool mk = (u >= 0.f) && (u <= 1407.f) && (vv >= 0.f) && (vv <= 511.f);
            if (mk) { cnt[s] += 1.f; viewany |= (1u << v); }
        }
    }
    float na = 0.f, ds_[SZ];
    for (int s = 0; s < SZ; ++s) {
        if (cnt[s] > 0.f) { na += 1.f; ds_[s] = 1.0f / fmaxf(cnt[s], 1e-6f); }
        else ds_[s] = 0.f;
    }
    float dna = 1.0f / fmaxf(na, 1e-6f);

    float ua = 0.f, va = 0.f, za = 0.f;
    for (int v = 0; v < VC; ++v) {
        const float* M = sM + v * 16;
        for (int s = 0; s < SZ; ++s) {
            float z3 = sZ[s];
            float c0 = M[0]*bx + M[1]*by + M[2]*z3  + M[3];
            float c1 = M[4]*bx + M[5]*by + M[6]*z3  + M[7];
            float c2 = M[8]*bx + M[9]*by + M[10]*z3 + M[11];
            float zz = fmaxf(c2, 1e-6f);
            float u = c0 / zz, vv = c1 / zz;
            bool mk = (u >= 0.f) && (u <= 1407.f) && (vv >= 0.f) && (vv <= 511.f);
            float coef = mk ? ds_[s] * dna : 0.f;
            ua += coef * u; va += coef * vv; za += coef * zz;
            float ix = fminf(fmaxf(u * 0.0625f, -8.f), 1.0e4f);   // u/16
            float iy = fminf(fmaxf(vv * 0.0625f, -8.f), 1.0e4f);  // v/16
            float x0f = floorf(ix), y0f = floorf(iy);
            float wx1 = ix - x0f, wy1 = iy - y0f;
            int x0 = (int)x0f, y0 = (int)y0f;
            int   cx[4] = {x0, x0 + 1, x0, x0 + 1};
            int   cy[4] = {y0, y0, y0 + 1, y0 + 1};
            float cw[4] = {(1.f - wy1) * (1.f - wx1), (1.f - wy1) * wx1,
                           wy1 * (1.f - wx1),         wy1 * wx1};
            int e0 = (v * SZ + s) * 4;
#pragma unroll
            for (int q = 0; q < 4; ++q) {
                bool valid = (cx[q] >= 0) && (cx[q] <= WPV - 1) &&
                             (cy[q] >= 0) && (cy[q] <= HPV - 1);
                int xi = cx[q] < 0 ? 0 : (cx[q] > WPV - 1 ? WPV - 1 : cx[q]);
                int yi = cy[q] < 0 ? 0 : (cy[q] > HPV - 1 ? HPV - 1 : cy[q]);
                midx[(size_t)l * 96 + e0 + q] = v * PATCH + yi * WPV + xi;
                mw  [(size_t)l * 96 + e0 + q] = valid ? cw[q] * coef : 0.f;
            }
        }
    }
    float vr = fminf((float)__popc(viewany) * (1.0f / 6.0f), 1.0f);
    posf[0 * L1V + l] = fminf(fmaxf(ua * (1.0f / 1407.0f), 0.f), 1.f);
    posf[1 * L1V + l] = fminf(fmaxf(va * (1.0f / 511.0f),  0.f), 1.f);
    posf[2 * L1V + l] = vr;
    posf[3 * L1V + l] = fminf(fmaxf((za - 1e-6f) * (1.0f / (3.0f - 1e-6f)), 0.f), 1.f);
}

// ---------------- DINO weighted gather: dmT[l, c] (coalesced stores) --------

__global__ __launch_bounds__(256) void gather_k(const float* __restrict__ feat,
                                                const int* __restrict__ midx,
                                                const float* __restrict__ mw,
                                                f16* __restrict__ dmT) {
    int l = blockIdx.x;
    int t = threadIdx.x;
    __shared__ int   sI[96];
    __shared__ float sW[96];
    if (t < 96) { sI[t] = midx[(size_t)l * 96 + t]; sW[t] = mw[(size_t)l * 96 + t]; }
    __syncthreads();
    float a0 = 0.f, a1 = 0.f, a2 = 0.f;
    for (int e = 0; e < 96; ++e) {
        float w = sW[e];
        if (w != 0.f) {
            const float* fb = feat + (size_t)sI[e] * CDINO + t;
            __builtin_prefetch(fb, 0, 1);
            a0 += w * fb[0];
            a1 += w * fb[256];
            a2 += w * fb[512];
        }
    }
    dmT[(size_t)l * CDINO + t      ] = (f16)a0;
    dmT[(size_t)l * CDINO + t + 256] = (f16)a1;
    dmT[(size_t)l * CDINO + t + 512] = (f16)a2;
}

// ---------------- tiny pos-MLP first layers (K=2 / K=4) + exact GELU --------
// one block per BEV cell; threads = output channel; L-major coalesced stores

__global__ __launch_bounds__(256) void posmlp_self_k(const float* __restrict__ w1,
                                                     const float* __restrict__ b1,
                                                     f16* __restrict__ outT) {
    int l = blockIdx.x, o = threadIdx.x;
    int i = l / 50, j = l % 50;
    float xn = j * (1.0f / 49.0f), yn = i * (1.0f / 49.0f);
    float a = w1[o * 2] * xn + w1[o * 2 + 1] * yn + b1[o];
    outT[(size_t)l * 256 + o] = (f16)gelu_exact(a);
}

__global__ __launch_bounds__(256) void posmlp_dino_k(const float* __restrict__ w1,
                                                     const float* __restrict__ b1,
                                                     const float* __restrict__ pf,
                                                     f16* __restrict__ outT) {
    int l = blockIdx.x, o = threadIdx.x;
    float a = w1[o*4] * pf[l] + w1[o*4+1] * pf[L1V + l] +
              w1[o*4+2] * pf[2*L1V + l] + w1[o*4+3] * pf[3*L1V + l] + b1[o];
    outT[(size_t)l * 256 + o] = (f16)gelu_exact(a);
}

// ---------------- WMMA GEMM: Y = W(CoutxCin) * X^T(ncols x ldb) + b ---------
// B is L-major: B[k][n] = Bt[n*ldb + k] -> contiguous-K fragment loads.
// One wave per 64x16 output strip (MT=4 m-tiles share each B fragment).

template<int MT>
__global__ __launch_bounds__(32) void gemm_k(
    const f16* __restrict__ A, const f16* __restrict__ Bt, int ldb,
    const float* __restrict__ bias, int Cin,
    f16* __restrict__ o16, int ld16, int co16, int tr16,
    float sc, const float* __restrict__ scp,
    float* __restrict__ o32, int ld32, int nv32, const float* __restrict__ resid) {
    const int lane = threadIdx.x & 31;
    const int ln = lane & 15, h2 = lane >> 4;
    const int mb = blockIdx.x * (16 * MT);
    const int n0 = blockIdx.y * 16;
    v8f acc[MT];
#pragma unroll
    for (int t = 0; t < MT; ++t) acc[t] = (v8f){};
    const f16* Bp = Bt + (size_t)(n0 + ln) * ldb;
    for (int k0 = 0; k0 < Cin; k0 += 32) {
        v16h b = frag_b_nm(Bp + k0, h2);
#pragma unroll
        for (int t = 0; t < MT; ++t) {
            v16h a = frag_a_rm(A + (size_t)(mb + t * 16 + ln) * Cin + k0, h2);
            acc[t] = __builtin_amdgcn_wmma_f32_16x16x32_f16(false, a, false, b,
                                                            (short)0, acc[t],
                                                            false, false);
        }
    }
    float s = sc * (scp ? scp[0] : 1.0f);
#pragma unroll
    for (int t = 0; t < MT; ++t) {
#pragma unroll
        for (int r = 0; r < 8; ++r) {
            int row = mb + t * 16 + r + 8 * h2;
            float val = acc[t][r] + bias[row];
            if (o16) {
                if (tr16)   // L-major: 8 contiguous f16 per lane per tile
                    o16[(size_t)(co16 + n0 + ln) * ld16 + row] = (f16)(val * s);
                else
                    o16[(size_t)row * ld16 + co16 + n0 + ln] = (f16)(val * s);
            }
            if (o32) {
                int col = n0 + ln;
                if (col < nv32) {
                    float rv = resid ? resid[(size_t)row * ld32 + col] : 0.0f;
                    o32[(size_t)row * ld32 + col] = val + rv;
                }
            }
        }
    }
}

// ---------------- flash attention: 1 wave per (head, 16-query tile) ---------
// K (L-major) and V (c-major) 32x32 tiles staged into LDS by the Tensor Data
// Mover (double buffered, s_wait_tensorcnt); fragments read as ds b128.

__global__ __launch_bounds__(32) void attn_k(const f16* __restrict__ Qt,
                                             const f16* __restrict__ Kt,
                                             const f16* __restrict__ V,
                                             f16* __restrict__ Ot) {
    const int qt = blockIdx.x;
    const int h  = blockIdx.y;
    const int lane = threadIdx.x & 31;
    const int ln = lane & 15, h2 = lane >> 4;

    __shared__ float s_logit[16][32];
    __shared__ f16   s_prob[16][32];
    __shared__ float s_m[16], s_l[16], s_alpha[16], s_linv[16];
#if HAVE_TDM
    __shared__ f16 sK[2][32 * 32];   // [buf][key*32 + k]  (L-major tile)
    __shared__ f16 sV[2][32 * 32];   // [buf][c*32 + key]
#endif

    const f16* Kbase = Kt + (size_t)h * DKH;                 // + s*256 per key
    const f16* Vbase = V + (size_t)(h * DKH) * LKP;          // + s per key

    // Q fragment: A[m][k] = Qt[(qt*16+m)*256 + h*32 + k]  (contiguous K)
    v16h aq = frag_a_rm(Qt + (size_t)(qt * 16 + ln) * 256 + h * DKH, h2);

    if (lane < 16) { s_m[lane] = -1e30f; s_l[lane] = 0.0f; }
    v8f c0 = {}, c1 = {};
    const v8f zf = {};

#if HAVE_TDM
    // prologue: stage tile 0 into buffer 0
    tdm_load_tile_f16((unsigned)(size_t)&sK[0][0], Kbase, 32, 32, 256);
    tdm_load_tile_f16((unsigned)(size_t)&sV[0][0], Vbase, 32, 32, LKP);
#endif
    __syncthreads();

    for (int kt = 0; kt < KTILES; ++kt) {
        const int s0 = kt * 32;
#if HAVE_TDM
        const int buf = kt & 1;
        if (kt + 1 < KTILES) {
            const int nb = (kt + 1) & 1;
            tdm_load_tile_f16((unsigned)(size_t)&sK[nb][0],
                              Kbase + (size_t)(s0 + 32) * 256, 32, 32, 256);
            tdm_load_tile_f16((unsigned)(size_t)&sV[nb][0],
                              Vbase + s0 + 32, 32, 32, LKP);
            __builtin_amdgcn_s_wait_tensorcnt(2);   // current tile's 2 ops done
        } else {
            __builtin_amdgcn_s_wait_tensorcnt(0);
        }
        asm volatile("" ::: "memory");
        // B[k][n] = sK[n*32 + k]: contiguous per lane -> ds b128
        v16h bk0, bk1;
#pragma unroll
        for (int j = 0; j < 8; ++j) {
            int K0 = 16 * h2 + 2 * j;
            bk0[2*j]   = sK[buf][ln * 32 + K0];
            bk0[2*j+1] = sK[buf][ln * 32 + K0 + 1];
            bk1[2*j]   = sK[buf][(16 + ln) * 32 + K0];
            bk1[2*j+1] = sK[buf][(16 + ln) * 32 + K0 + 1];
        }
#else
        v16h bk0 = frag_b_nm(Kbase + (size_t)(s0 + ln) * 256, h2);
        v16h bk1 = frag_b_nm(Kbase + (size_t)(s0 + 16 + ln) * 256, h2);
#endif
        v8f l0 = __builtin_amdgcn_wmma_f32_16x16x32_f16(false, aq, false, bk0,
                                                        (short)0, zf, false, false);
        v8f l1 = __builtin_amdgcn_wmma_f32_16x16x32_f16(false, aq, false, bk1,
                                                        (short)0, zf, false, false);
#pragma unroll
        for (int r = 0; r < 8; ++r) {
            int row = r + 8 * h2;
            s_logit[row][ln]      = (s0 + ln      < LKVAL) ? l0[r] : -1e30f;
            s_logit[row][16 + ln] = (s0 + 16 + ln < LKVAL) ? l1[r] : -1e30f;
        }
        __syncthreads();
        if (lane < 16) {                      // per-query online softmax stats
            float mo = s_m[lane], mx = mo;
#pragma unroll 4
            for (int k = 0; k < 32; ++k) mx = fmaxf(mx, s_logit[lane][k]);
            float al = __expf(mo - mx);
            float lsum = s_l[lane] * al;
#pragma unroll 4
            for (int k = 0; k < 32; ++k) {
                float p = __expf(s_logit[lane][k] - mx);
                s_prob[lane][k] = (f16)p;
                lsum += p;
            }
            s_m[lane] = mx; s_l[lane] = lsum; s_alpha[lane] = al;
        }
        __syncthreads();
#pragma unroll
        for (int r = 0; r < 8; ++r) {
            float al = s_alpha[r + 8 * h2];
            c0[r] *= al; c1[r] *= al;
        }
        v16h ap;
#pragma unroll
        for (int j = 0; j < 8; ++j) {
            int K0 = 16 * h2 + 2 * j;
            ap[2*j]   = s_prob[ln][K0];
            ap[2*j+1] = s_prob[ln][K0+1];
        }
#if HAVE_TDM
        v16h bv0, bv1;
#pragma unroll
        for (int j = 0; j < 8; ++j) {
            int K0 = 16 * h2 + 2 * j;
            bv0[2*j]   = sV[buf][ln * 32 + K0];
            bv0[2*j+1] = sV[buf][ln * 32 + K0 + 1];
            bv1[2*j]   = sV[buf][(16 + ln) * 32 + K0];
            bv1[2*j+1] = sV[buf][(16 + ln) * 32 + K0 + 1];
        }
#else
        v16h bv0 = frag_b_nm(Vbase + (size_t)ln        * LKP + s0, h2);
        v16h bv1 = frag_b_nm(Vbase + (size_t)(16 + ln) * LKP + s0, h2);
#endif
        c0 = __builtin_amdgcn_wmma_f32_16x16x32_f16(false, ap, false, bv0,
                                                    (short)0, c0, false, false);
        c1 = __builtin_amdgcn_wmma_f32_16x16x32_f16(false, ap, false, bv1,
                                                    (short)0, c1, false, false);
        __syncthreads();
    }
    if (lane < 16) s_linv[lane] = 1.0f / fmaxf(s_l[lane], 1e-30f);
    __syncthreads();
#pragma unroll
    for (int r = 0; r < 8; ++r) {
        int row = r + 8 * h2;                  // query within tile
        float inv = s_linv[row];
        // Ot is L-major (LQP x 256); lanes 0..15 write 16 contiguous halves
        Ot[(size_t)(qt * 16 + row) * 256 + h * DKH + ln]      = (f16)(c0[r] * inv);
        Ot[(size_t)(qt * 16 + row) * 256 + h * DKH + 16 + ln] = (f16)(c1[r] * inv);
    }
}

// ---------------- host orchestration ----------------------------------------

extern "C" void kernel_launch(void* const* d_in, const int* in_sizes, int n_in,
                              void* d_out, int out_size, void* d_ws, size_t ws_size,
                              hipStream_t stream) {
    (void)in_sizes; (void)n_in; (void)out_size; (void)ws_size;
    const float* bev  = (const float*)d_in[0];
    const float* toks = (const float*)d_in[1];
    const float* l2i  = (const float*)d_in[2];
    const float* zsmp = (const float*)d_in[3];
    const float* gq   = (const float*)d_in[4];
    const float* gqb  = (const float*)d_in[5];
    const float* qw   = (const float*)d_in[6];
    const float* qb   = (const float*)d_in[7];
    const float* sm1w = (const float*)d_in[8];
    const float* sm1b = (const float*)d_in[9];
    const float* sm2w = (const float*)d_in[10];
    const float* sm2b = (const float*)d_in[11];
    const float* wksw = (const float*)d_in[12];
    const float* wksb = (const float*)d_in[13];
    const float* wvsw = (const float*)d_in[14];
    const float* wvsb = (const float*)d_in[15];
    const float* dm1w = (const float*)d_in[16];
    const float* dm1b = (const float*)d_in[17];
    const float* dm2w = (const float*)d_in[18];
    const float* dm2b = (const float*)d_in[19];
    const float* dcw  = (const float*)d_in[20];
    const float* dcb  = (const float*)d_in[21];
    const float* wkdw = (const float*)d_in[22];
    const float* wkdb = (const float*)d_in[23];
    const float* wvdw = (const float*)d_in[24];
    const float* wvdb = (const float*)d_in[25];
    const float* pwo  = (const float*)d_in[26];
    const float* pob  = (const float*)d_in[27];
    const float* gate = (const float*)d_in[28];
    float* out = (float*)d_out;

    char* ws = (char*)d_ws;
    size_t off = 0;
    auto alloc = [&](size_t bytes) -> void* {
        void* p = ws + off;
        off += (bytes + 255) & ~(size_t)255;
        return p;
    };
    f16* wq   = (f16*)alloc((size_t)65536 * 2);
    f16* wsm2 = (f16*)alloc((size_t)65536 * 2);
    f16* wks  = (f16*)alloc((size_t)131072 * 2);
    f16* wvs  = (f16*)alloc((size_t)65536 * 2);
    f16* wdm2 = (f16*)alloc((size_t)65536 * 2);
    f16* wdc  = (f16*)alloc((size_t)196608 * 2);
    f16* wkd  = (f16*)alloc((size_t)131072 * 2);
    f16* wvd  = (f16*)alloc((size_t)65536 * 2);
    f16* wpo  = (f16*)alloc((size_t)65536 * 2);
    const size_t ACT_ELEMS =
        (size_t)LQP * (512 + 512 + 256 + 256 + 256 + 256 + 768 + 256) +
        (size_t)LKP * 512;
    f16* actbase  = (f16*)alloc(ACT_ELEMS * 2);
    f16* selfcatT = actbase;                               // LQP x 512 (x||pos)
    f16* dinocatT = selfcatT + (size_t)LQP * 512;          // LQP x 512
    f16* xnT      = dinocatT + (size_t)LQP * 512;          // LQP x 256
    f16* pos1sT   = xnT      + (size_t)LQP * 256;          // LQP x 256
    f16* pos1dT   = pos1sT   + (size_t)LQP * 256;          // LQP x 256
    f16* q16T     = pos1dT   + (size_t)LQP * 256;          // LQP x 256 (scaled)
    f16* dmT      = q16T     + (size_t)LQP * 256;          // LQP x 768
    f16* aoT      = dmT      + (size_t)LQP * 768;          // LQP x 256
    f16* kmixT    = aoT      + (size_t)LQP * 256;          // LKP x 256 (scaled)
    f16* vmix     = kmixT    + (size_t)LKP * 256;          // 256 x LKP (gated)
    float* posf = (float*)alloc((size_t)4 * L1V * 4);
    int*   midx = (int*)alloc((size_t)L1V * 96 * 4);
    float* mw   = (float*)alloc((size_t)L1V * 96 * 4);
    float* gbuf = (float*)alloc(256);

    dim3 b256(256);
    { int n = (int)ACT_ELEMS;
      zero16_k<<<dim3((n + 255) / 256), b256, 0, stream>>>(actbase, n); }
    sigm_k<<<dim3(1), dim3(1), 0, stream>>>(gate, gbuf);
    auto cvt = [&](const float* s, f16* d, int n) {
        cvt16_k<<<dim3((n + 255) / 256), b256, 0, stream>>>(s, d, n);
    };
    cvt(qw, wq, 65536);     cvt(sm2w, wsm2, 65536); cvt(wksw, wks, 131072);
    cvt(wvsw, wvs, 65536);  cvt(dm2w, wdm2, 65536); cvt(dcw, wdc, 196608);
    cvt(wkdw, wkd, 131072); cvt(wvdw, wvd, 65536);  cvt(pwo, wpo, 65536);

    xpose_cvt_k<<<dim3(L1V), b256, 0, stream>>>(bev, selfcatT, 512);
    gn_k<<<dim3(32), b256, 0, stream>>>(bev, gq, gqb, xnT);
    geom_k<<<dim3((L1V + 255) / 256), b256, 0, stream>>>(l2i, zsmp, midx, mw, posf);
    posmlp_self_k<<<dim3(L1V), b256, 0, stream>>>(sm1w, sm1b, pos1sT);
    gather_k<<<dim3(L1V), b256, 0, stream>>>(toks, midx, mw, dmT);
    posmlp_dino_k<<<dim3(L1V), b256, 0, stream>>>(dm1w, dm1b, posf, pos1dT);

    auto gemm = [&](const f16* A, const f16* Bt, int ldb, const float* bias, int Cin,
                    f16* o16, int ld16, int co, int tr, float sc, const float* scp,
                    float* o32, int ld32, int nv, const float* resid) {
        gemm_k<4><<<dim3(256 / 64, LQP / 16), dim3(32), 0, stream>>>(
            A, Bt, ldb, bias, Cin, o16, ld16, co, tr, sc, scp, o32, ld32, nv, resid);
    };
    // self branch
    gemm(wsm2, pos1sT, 256, sm2b, 256, selfcatT + 256, 512, 0, 1, 1.f, nullptr,
         nullptr, 0, 0, nullptr);                                   // self_pos
    gemm(wq, xnT, 256, qb, 256, q16T, 256, 0, 1, SCALE_QK, nullptr,
         nullptr, 0, 0, nullptr);                                   // Q (scaled)
    gemm(wks, selfcatT, 512, wksb, 512, kmixT, 256, 0, 1, SCALE_QK, nullptr,
         nullptr, 0, 0, nullptr);                                   // K_self
    gemm(wvs, selfcatT, 512, wvsb, 256, vmix, LKP, 0, 0, 1.f, nullptr,
         nullptr, 0, 0, nullptr);                                   // V_self
    // dino branch (after K_self/V_self so dino rows overwrite pad spill)
    gemm(wdc, dmT, 768, dcb, 768, dinocatT, 512, 0, 1, 1.f, nullptr,
         nullptr, 0, 0, nullptr);                                   // dino_c
    gemm(wdm2, pos1dT, 256, dm2b, 256, dinocatT + 256, 512, 0, 1, 1.f, nullptr,
         nullptr, 0, 0, nullptr);                                   // dino_pos
    gemm(wkd, dinocatT, 512, wkdb, 512, kmixT, 256, 2500, 1, SCALE_QK, nullptr,
         nullptr, 0, 0, nullptr);                                   // K_dino
    gemm(wvd, dinocatT, 512, wvdb, 256, vmix, LKP, 2500, 0, 1.f, gbuf,
         nullptr, 0, 0, nullptr);                                   // g * V_dino
    // flash attention
    attn_k<<<dim3(LQP / 16, NHEAD), dim3(32), 0, stream>>>(q16T, kmixT, vmix, aoT);
    // out = x + proj_out(attn)
    gemm(wpo, aoT, 256, pob, 256, nullptr, 0, 0, 0, 1.f, nullptr,
         out, L1V, L1V, bev);
}
